// AdvancedMoEDecoderBlock_10883447128125
// MI455X (gfx1250) — matrix operand order repaired and checked by reading, not compiled, not run on hardware
//
#include <hip/hip_runtime.h>

// ---------------- problem constants ----------------
#define DMODEL 1024
#define NH 16
#define NKV 4
#define HD 64
#define KVD 256
#define NE 8
#define LR 16
#define NHID 4096
#define SEQ 1024
#define NB 2
#define NT 2048          // B*S tokens
#define QKVW 1536        // D + KV + KV
#define LORA_SCALE 2.0f  // 32/16

// LDS element strides (multiples of 8 elems => 16B-aligned rows)
#define SA_LD 40   // A tile rows: 32 k + 8 pad
#define SBT_LD 40  // B tile stored transposed: [col][32 k + 8 pad]
#define KLD 72     // K tile rows: 64 dims + 8 pad
#define VT_LD 40   // V tile transposed: [dim][32 keys + 8 pad]

typedef __attribute__((ext_vector_type(16))) __bf16 v16bf;
typedef __attribute__((ext_vector_type(8)))  float  v8f;
typedef unsigned short us;

struct U32x8 { uint4 a, b; };   // 32 bytes, bit_cast target for v16bf

__device__ __forceinline__ us    f2bfbits(float f) { return __builtin_bit_cast(us, (__bf16)f); }
__device__ __forceinline__ float bf2f(us x)        { return (float)__builtin_bit_cast(__bf16, x); }
__device__ __forceinline__ unsigned pack2bf(float lo, float hi) {
  return (unsigned)f2bfbits(lo) | ((unsigned)f2bfbits(hi) << 16);
}
__device__ __forceinline__ v16bf mkfrag(uint4 a, uint4 b) {
  U32x8 t{a, b};
  return __builtin_bit_cast(v16bf, t);
}

__device__ __forceinline__ v8f wmma_b(v16bf a, v16bf b, v8f c) {
  return __builtin_amdgcn_wmma_f32_16x16x32_bf16(false, a, false, b, (short)0, c, false, false);
}

// A fragment: 16x32 bf16 tile (row-major, stride SA_LD), two b128 LDS loads
__device__ __forceinline__ v16bf frag_a(const __bf16* sm, int lane) {
  int row = lane & 15;
  int kb  = (lane < 16) ? 0 : 16;
  const uint4* p = (const uint4*)(sm + row * SA_LD + kb);        // k = kb..kb+7
  const uint4* q = (const uint4*)(sm + row * SA_LD + kb + 8);    // k = kb+8..kb+15
  return mkfrag(p[0], q[0]);
}
// B fragment from transposed LDS tile [col][k]: contiguous 16 elems -> two b128 loads
__device__ __forceinline__ v16bf frag_bT(const __bf16* smT, int lane) {
  int col = lane & 15;
  int kb  = (lane < 16) ? 0 : 16;
  const uint4* p = (const uint4*)(smT + col * SBT_LD + kb);
  return mkfrag(p[0], p[1]);
}

// stage 128x32 bf16 A tile: two uint4 loads + two b128 LDS stores per thread
__device__ __forceinline__ void stageA128(__bf16* sA, const us* A, int lda, int m0, int kk,
                                          const int* permE, int Mbound, int tid) {
  int r   = tid >> 1;          // 0..127
  int c16 = (tid & 1) * 16;    // 0 or 16
  uint4 v0 = {0u, 0u, 0u, 0u}, v1 = {0u, 0u, 0u, 0u};
  int row = m0 + r;
  if (row < Mbound) {
    int grow = permE ? permE[row] : row;
    const uint4* src = (const uint4*)(A + (size_t)grow * lda + kk + c16);
    v0 = src[0];
    v1 = src[1];
  }
  uint4* dst = (uint4*)(sA + r * SA_LD + c16);
  dst[0] = v0;
  dst[1] = v1;
}
// stage 32x64 fp32 B tile transposed into LDS as bf16: two float4 loads + 4 b32 stores
__device__ __forceinline__ void stageBT(__bf16* sBt, const float* Bm, int ldb, int n0, int kk, int tid) {
  int r0 = (tid >> 4) * 2;    // even k row 0..30
  int c4 = (tid & 15) * 4;    // col group 0..60
  const float* p0 = Bm + (size_t)(kk + r0) * ldb + n0 + c4;
  float4 a = *(const float4*)p0;
  float4 b = *(const float4*)(p0 + ldb);
  *(unsigned*)(sBt + (c4 + 0) * SBT_LD + r0) = pack2bf(a.x, b.x);
  *(unsigned*)(sBt + (c4 + 1) * SBT_LD + r0) = pack2bf(a.y, b.y);
  *(unsigned*)(sBt + (c4 + 2) * SBT_LD + r0) = pack2bf(a.z, b.z);
  *(unsigned*)(sBt + (c4 + 3) * SBT_LD + r0) = pack2bf(a.w, b.w);
}

// one A-frag feeds the whole 64-wide strip: 4 WMMAs per K-step
__device__ __forceinline__ void mac4(v16bf af, const __bf16* sBt, int lane,
                                     v8f& a0, v8f& a1, v8f& a2, v8f& a3) {
  a0 = wmma_b(af, frag_bT(sBt + 0  * SBT_LD, lane), a0);
  a1 = wmma_b(af, frag_bT(sBt + 16 * SBT_LD, lane), a1);
  a2 = wmma_b(af, frag_bT(sBt + 32 * SBT_LD, lane), a2);
  a3 = wmma_b(af, frag_bT(sBt + 48 * SBT_LD, lane), a3);
}

// ---------------- LoRA-folded effective weights ----------------
__global__ __launch_bounds__(256) void k_prep_weff(
    const float* WqT, const float* WkT, const float* WvT,
    const float* qA, const float* qB, const float* kA, const float* kB,
    const float* vA, const float* vB, float* Weff) {
  int idx = blockIdx.x * 256 + threadIdx.x;       // 1024*1536
  int i = idx / QKVW, j = idx - i * QKVW;
  float acc;
  if (j < DMODEL) {
    acc = WqT[(size_t)i * DMODEL + j];
#pragma unroll
    for (int r = 0; r < LR; ++r) acc += LORA_SCALE * qA[i * LR + r] * qB[r * DMODEL + j];
  } else if (j < DMODEL + KVD) {
    int jj = j - DMODEL;
    acc = WkT[(size_t)i * KVD + jj];
#pragma unroll
    for (int r = 0; r < LR; ++r) acc += LORA_SCALE * kA[i * LR + r] * kB[r * KVD + jj];
  } else {
    int jj = j - DMODEL - KVD;
    acc = WvT[(size_t)i * KVD + jj];
#pragma unroll
    for (int r = 0; r < LR; ++r) acc += LORA_SCALE * vA[i * LR + r] * vB[r * KVD + jj];
  }
  Weff[idx] = acc;
}

__global__ __launch_bounds__(256) void k_prep_geff(
    const float* gateWT, const float* gA, const float* gB, float* geff) {
  int idx = blockIdx.x * 256 + threadIdx.x;       // 1024*8
  int i = idx >> 3, e = idx & 7;
  float acc = gateWT[i * NE + e];
#pragma unroll
  for (int r = 0; r < LR; ++r) acc += LORA_SCALE * gA[i * LR + r] * gB[r * NE + e];
  geff[idx] = acc;
}

// ---------------- RMSNorm (+optional task embedding) -> bf16 ----------------
__global__ __launch_bounds__(256) void k_rmsnorm(const float* xin, const float* wgt,
                                                 const float* temb, const int* tids, us* outbf) {
  int t = blockIdx.x;
  __shared__ float red[8];
  const float* row = xin + (size_t)t * DMODEL;
  float ss = 0.f;
  for (int i = threadIdx.x; i < DMODEL; i += 256) { float v = row[i]; ss += v * v; }
  ss += __shfl_xor(ss, 16, 32); ss += __shfl_xor(ss, 8, 32); ss += __shfl_xor(ss, 4, 32);
  ss += __shfl_xor(ss, 2, 32);  ss += __shfl_xor(ss, 1, 32);
  if ((threadIdx.x & 31) == 0) red[threadIdx.x >> 5] = ss;
  __syncthreads();
  float tot = 0.f;
#pragma unroll
  for (int k = 0; k < 8; ++k) tot += red[k];
  float sc = rsqrtf(tot * (1.0f / DMODEL) + 1e-6f);
  const float* te = temb ? (temb + (size_t)tids[t] * DMODEL) : nullptr;
  for (int i = threadIdx.x; i < DMODEL; i += 256) {
    float v = row[i] * sc * wgt[i];
    if (te) v += te[i];
    outbf[(size_t)t * DMODEL + i] = f2bfbits(v);
  }
}

// ---------------- generic 128x64-tile WMMA GEMM: C = A_bf16 @ B_f32 (+addsrc) ----------------
__global__ __launch_bounds__(256) void k_gemm_plain(
    const us* A, int lda, const float* Bm, int ldb,
    const float* addsrc, float* Cf, us* Cbf, int ldc, int Kdim) {
  __shared__ __bf16 sA[128 * SA_LD];
  __shared__ __bf16 sBt[64 * SBT_LD];
  int tid = threadIdx.x, lane = tid & 31, w = tid >> 5;   // wave w owns rows [w*16, w*16+16)
  int m0 = blockIdx.y * 128, n0 = blockIdx.x * 64;
  v8f a0 = {}, a1 = {}, a2 = {}, a3 = {};
  for (int kk = 0; kk < Kdim; kk += 32) {
    stageA128(sA, A, lda, m0, kk, nullptr, 1 << 30, tid);
    stageBT(sBt, Bm, ldb, n0, kk, tid);
    if (kk + 32 < Kdim)
      __builtin_prefetch(Bm + (size_t)(kk + 32 + ((tid >> 4) * 2)) * ldb + n0, 0, 1);
    __syncthreads();
    v16bf af = frag_a(sA + w * 16 * SA_LD, lane);
    mac4(af, sBt, lane, a0, a1, a2, a3);
    __syncthreads();
  }
  v8f accs[4] = {a0, a1, a2, a3};
#pragma unroll
  for (int t = 0; t < 4; ++t) {
#pragma unroll
    for (int i = 0; i < 8; ++i) {
      int rl = w * 16 + i + ((lane >> 4) << 3);
      int cl = t * 16 + (lane & 15);
      size_t row = m0 + rl, col = n0 + cl;
      float v = accs[t][i];
      if (addsrc) v += addsrc[row * ldc + col];
      if (Cf) Cf[row * ldc + col] = v;
      else    Cbf[row * ldc + col] = f2bfbits(v);
    }
  }
}

// ---------------- RoPE + GQA-layout repack -> bf16 ----------------
__global__ __launch_bounds__(256) void k_rope(const float* QKV, const float* fcos, const float* fsin,
                                              us* Qr, us* Kr, us* Vr) {
  int t = blockIdx.x;
  int b = t / SEQ, s = t % SEQ;
  const float* row = QKV + (size_t)t * QKVW;
  for (int i = threadIdx.x; i < DMODEL; i += 256) {
    int h = i >> 6, d = i & 63;
    float v = row[i];
    float other = row[(h << 6) + ((d < 32) ? d + 32 : d - 32)];
    float rot = (d < 32) ? -other : other;
    float out = v * fcos[s * HD + d] + rot * fsin[s * HD + d];
    Qr[((size_t)(b * NH + h) * SEQ + s) * HD + d] = f2bfbits(out);
  }
  for (int i = threadIdx.x; i < KVD; i += 256) {
    int h = i >> 6, d = i & 63;
    float v = row[DMODEL + i];
    float other = row[DMODEL + (h << 6) + ((d < 32) ? d + 32 : d - 32)];
    float rot = (d < 32) ? -other : other;
    float out = v * fcos[s * HD + d] + rot * fsin[s * HD + d];
    Kr[((size_t)(b * NKV + h) * SEQ + s) * HD + d] = f2bfbits(out);
    Vr[((size_t)(b * NKV + h) * SEQ + s) * HD + d] = f2bfbits(row[DMODEL + KVD + i]);
  }
}

// ---------------- flash attention (causal, GQA), WMMA QK^T and PV ----------------
__global__ __launch_bounds__(256) void k_flash(const us* Qr, const us* Kr, const us* Vr, us* ctx) {
  __shared__ __bf16 sK[32 * KLD];    // row-major [key][dim]
  __shared__ __bf16 sVt[64 * VT_LD]; // transposed [dim][key]
  __shared__ float  pstage[8 * 16 * 32];
  int tid = threadIdx.x, lane = tid & 31, w = tid >> 5;
  int b = blockIdx.z, hq = blockIdx.y, hkv = hq >> 2;
  int q0 = blockIdx.x * 128 + w * 16;
  // Q fragments (16 rows x 64 dims = two 16x32 A-frags), vectorized global loads
  v16bf qf0, qf1;
  {
    const us* qp = Qr + ((size_t)((b * NH + hq) * SEQ + q0 + (lane & 15))) * HD;
    int kb = (lane < 16) ? 0 : 16;
    uint4 u0 = *(const uint4*)(qp + kb);
    uint4 u1 = *(const uint4*)(qp + kb + 8);
    uint4 u2 = *(const uint4*)(qp + 32 + kb);
    uint4 u3 = *(const uint4*)(qp + 32 + kb + 8);
    qf0 = mkfrag(u0, u1);
    qf1 = mkfrag(u2, u3);
  }
  v8f o0 = {}, o1 = {}, o2 = {}, o3 = {};
  float mrow[8], lrow[8];
#pragma unroll
  for (int i = 0; i < 8; ++i) { mrow[i] = -1e30f; lrow[i] = 0.f; }
  float* Pw = pstage + w * (16 * 32);
  int jmax = blockIdx.x * 128 + 128;
  for (int j = 0; j < jmax; j += 32) {
    const us* kg = Kr + ((size_t)((b * NKV + hkv) * SEQ + j)) * HD;
    const us* vg = Vr + ((size_t)((b * NKV + hkv) * SEQ + j)) * HD;
    {
      // K: straight b128 copy (row-major)
      int r = tid >> 3, c8 = (tid & 7) * 8;
      *(uint4*)(sK + r * KLD + c8) = *(const uint4*)(kg + r * HD + c8);
      // V: transpose-stage (pack key pairs per dim)
      int r0 = (tid >> 4) * 2, c4 = (tid & 15) * 4;
      uint2 a = *(const uint2*)(vg + r0 * HD + c4);
      uint2 bb = *(const uint2*)(vg + (r0 + 1) * HD + c4);
      *(unsigned*)(sVt + (c4 + 0) * VT_LD + r0) = (a.x & 0xffffu) | (bb.x << 16);
      *(unsigned*)(sVt + (c4 + 1) * VT_LD + r0) = (a.x >> 16) | (bb.x & 0xffff0000u);
      *(unsigned*)(sVt + (c4 + 2) * VT_LD + r0) = (a.y & 0xffffu) | (bb.y << 16);
      *(unsigned*)(sVt + (c4 + 3) * VT_LD + r0) = (a.y >> 16) | (bb.y & 0xffff0000u);
    }
    __syncthreads();
    if (j <= q0 + 15) {              // wave-uniform -> EXEC all-ones inside
      v8f s0 = {}, s1 = {};
      {
        int kb = (lane < 16) ? 0 : 16;
        const __bf16* k0 = sK + (lane & 15) * KLD + kb;        // key tile 0
        const __bf16* k1 = sK + (16 + (lane & 15)) * KLD + kb; // key tile 1
        const uint4* p;
        p = (const uint4*)k0;        s0 = wmma_b(qf0, mkfrag(p[0], p[1]), s0);
        p = (const uint4*)(k0 + 32); s0 = wmma_b(qf1, mkfrag(p[0], p[1]), s0);
        p = (const uint4*)k1;        s1 = wmma_b(qf0, mkfrag(p[0], p[1]), s1);
        p = (const uint4*)(k1 + 32); s1 = wmma_b(qf1, mkfrag(p[0], p[1]), s1);
      }
      float st0[8], st1[8];
      int rbase = q0 + ((lane >> 4) << 3);
      int c0 = j + (lane & 15);
#pragma unroll
      for (int i = 0; i < 8; ++i) {
        int rr = rbase + i;
        st0[i] = (c0      <= rr) ? s0[i] * 0.125f : -1e30f;
        st1[i] = (c0 + 16 <= rr) ? s1[i] * 0.125f : -1e30f;
      }
#pragma unroll
      for (int i = 0; i < 8; ++i) {
        float mx = fmaxf(st0[i], st1[i]);
        mx = fmaxf(mx, __shfl_xor(mx, 1, 32));
        mx = fmaxf(mx, __shfl_xor(mx, 2, 32));
        mx = fmaxf(mx, __shfl_xor(mx, 4, 32));
        mx = fmaxf(mx, __shfl_xor(mx, 8, 32));
        float nm = fmaxf(mrow[i], mx);
        float sc = __expf(mrow[i] - nm);
        mrow[i] = nm;
        float p0 = __expf(st0[i] - nm);
        float p1 = __expf(st1[i] - nm);
        st0[i] = p0; st1[i] = p1;
        float rs = p0 + p1;
        rs += __shfl_xor(rs, 1, 32);
        rs += __shfl_xor(rs, 2, 32);
        rs += __shfl_xor(rs, 4, 32);
        rs += __shfl_xor(rs, 8, 32);
        lrow[i] = lrow[i] * sc + rs;
        o0[i] *= sc; o1[i] *= sc; o2[i] *= sc; o3[i] *= sc;
      }
      // C-layout -> A-layout via wave-private LDS (DS ops in-order per wave)
#pragma unroll
      for (int i = 0; i < 8; ++i) {
        int row = i + ((lane >> 4) << 3);
        Pw[row * 32 + (lane & 15)]      = st0[i];
        Pw[row * 32 + 16 + (lane & 15)] = st1[i];
      }
      v16bf pf;
      {
        int row = lane & 15;
        int kb = (lane < 16) ? 0 : 16;
        const float* pr = Pw + row * 32 + kb;
        float4 f0 = *(const float4*)(pr);
        float4 f1 = *(const float4*)(pr + 4);
        float4 f2 = *(const float4*)(pr + 8);
        float4 f3 = *(const float4*)(pr + 12);
        uint4 lo = {pack2bf(f0.x, f0.y), pack2bf(f0.z, f0.w), pack2bf(f1.x, f1.y), pack2bf(f1.z, f1.w)};
        uint4 hi = {pack2bf(f2.x, f2.y), pack2bf(f2.z, f2.w), pack2bf(f3.x, f3.y), pack2bf(f3.z, f3.w)};
        pf = mkfrag(lo, hi);
      }
      {
        int kb = (lane < 16) ? 0 : 16;
        int cl = lane & 15;
        const uint4* p;
        p = (const uint4*)(sVt + (cl)      * VT_LD + kb); o0 = wmma_b(pf, mkfrag(p[0], p[1]), o0);
        p = (const uint4*)(sVt + (16 + cl) * VT_LD + kb); o1 = wmma_b(pf, mkfrag(p[0], p[1]), o1);
        p = (const uint4*)(sVt + (32 + cl) * VT_LD + kb); o2 = wmma_b(pf, mkfrag(p[0], p[1]), o2);
        p = (const uint4*)(sVt + (48 + cl) * VT_LD + kb); o3 = wmma_b(pf, mkfrag(p[0], p[1]), o3);
      }
    }
    __syncthreads();
  }
#pragma unroll
  for (int i = 0; i < 8; ++i) {
    float inv = 1.0f / lrow[i];
    int rl = i + ((lane >> 4) << 3);
    size_t tok = (size_t)b * SEQ + q0 + rl;
    us* cp = ctx + tok * DMODEL + hq * HD;
    cp[0  + (lane & 15)] = f2bfbits(o0[i] * inv);
    cp[16 + (lane & 15)] = f2bfbits(o1[i] * inv);
    cp[32 + (lane & 15)] = f2bfbits(o2[i] * inv);
    cp[48 + (lane & 15)] = f2bfbits(o3[i] * inv);
  }
}

// ---------------- top-2 gate (LoRA-folded weights) ----------------
__global__ __launch_bounds__(256) void k_gate(const us* xf, const float* geff, int* eids, float* gwts) {
  int lane = threadIdx.x & 31, w = threadIdx.x >> 5;
  int t = blockIdx.x * 8 + w;
  float lg[NE];
#pragma unroll
  for (int e = 0; e < NE; ++e) lg[e] = 0.f;
  for (int d = lane; d < DMODEL; d += 32) {
    float xv = bf2f(xf[(size_t)t * DMODEL + d]);
#pragma unroll
    for (int e = 0; e < NE; ++e) lg[e] += xv * geff[d * NE + e];
  }
#pragma unroll
  for (int e = 0; e < NE; ++e) {
    lg[e] += __shfl_xor(lg[e], 16, 32);
    lg[e] += __shfl_xor(lg[e], 8, 32);
    lg[e] += __shfl_xor(lg[e], 4, 32);
    lg[e] += __shfl_xor(lg[e], 2, 32);
    lg[e] += __shfl_xor(lg[e], 1, 32);
  }
  if (lane == 0) {
    int i1 = 0; float v1 = lg[0];
#pragma unroll
    for (int e = 1; e < NE; ++e) if (lg[e] > v1) { v1 = lg[e]; i1 = e; }
    int i2 = -1; float v2 = -3.4e38f;
#pragma unroll
    for (int e = 0; e < NE; ++e) if (e != i1 && lg[e] > v2) { v2 = lg[e]; i2 = e; }
    float e2 = __expf(v2 - v1);
    float den = 1.0f + e2;
    eids[2 * t] = i1; eids[2 * t + 1] = i2;
    gwts[2 * t] = 1.0f / den; gwts[2 * t + 1] = e2 / den;
  }
}

// ---------------- deterministic per-expert token list (ballot scan) ----------------
__global__ void k_build_lists(const int* eids, const float* gwts,
                              int* perm, float* wslot, int* posOf, int* counts) {
  int e = blockIdx.x, lane = threadIdx.x;
  int cn = 0;
  for (int base = 0; base < NT; base += 32) {
    int t = base + lane;
    int k = -1;
    if (eids[2 * t] == e) k = 0;
    else if (eids[2 * t + 1] == e) k = 1;
    unsigned long long bal = __ballot(k >= 0);
    unsigned m32 = (unsigned)bal;
    int pre = __popc(m32 & ((1u << lane) - 1));
    if (k >= 0) {
      int pos = cn + pre;
      perm[e * NT + pos] = t;
      wslot[e * NT + pos] = gwts[2 * t + k];
      posOf[2 * t + k] = pos;
    }
    cn += __popc(m32);
  }
  if (lane == 0) counts[e] = cn;
}

__global__ void k_scan(const int* counts, int* off) {
  if (threadIdx.x == 0 && blockIdx.x == 0) {
    int a = 0;
    for (int e = 0; e < NE; ++e) { off[e] = a; a += counts[e]; }
    off[NE] = a;
  }
}

// ---------------- routed expert GEMM 1: h1 = gather(xf) @ W1 + b1 ----------------
__global__ __launch_bounds__(256) void k_gemm_e1(
    const us* xf, const float* e_W1, const float* e_b1,
    const int* perm, const int* cnt, const int* off, us* h1) {
  int e = blockIdx.z;
  int cnte = cnt[e];
  int m0 = blockIdx.y * 128;
  if (m0 >= cnte) return;
  __shared__ __bf16 sA[128 * SA_LD];
  __shared__ __bf16 sBt[64 * SBT_LD];
  int tid = threadIdx.x, lane = tid & 31, w = tid >> 5;
  int n0 = blockIdx.x * 64;
  const float* Bm = e_W1 + (size_t)e * DMODEL * NHID;
  const int* permE = perm + e * NT;
  v8f a0 = {}, a1 = {}, a2 = {}, a3 = {};
  for (int kk = 0; kk < DMODEL; kk += 32) {
    stageA128(sA, xf, DMODEL, m0, kk, permE, cnte, tid);
    stageBT(sBt, Bm, NHID, n0, kk, tid);
    if (kk + 32 < DMODEL)
      __builtin_prefetch(Bm + (size_t)(kk + 32 + ((tid >> 4) * 2)) * NHID + n0, 0, 1);
    __syncthreads();
    v16bf af = frag_a(sA + w * 16 * SA_LD, lane);
    mac4(af, sBt, lane, a0, a1, a2, a3);
    __syncthreads();
  }
  int base = off[e];
  v8f accs[4] = {a0, a1, a2, a3};
#pragma unroll
  for (int t = 0; t < 4; ++t) {
#pragma unroll
    for (int i = 0; i < 8; ++i) {
      int rl = w * 16 + i + ((lane >> 4) << 3);
      int cl = t * 16 + (lane & 15);
      int slot = m0 + rl;
      if (slot < cnte) {
        int gcol = n0 + cl;
        h1[(size_t)(base + slot) * NHID + gcol] = f2bfbits(accs[t][i] + e_b1[e * NHID + gcol]);
      }
    }
  }
}

// ---------------- routed expert GEMM 2: contrib = w * silu(h1@Wg+bg) * (h1@Wv+bv) ----------------
__global__ __launch_bounds__(256) void k_gemm_e2(
    const us* h1, const float* e_Wg, const float* e_bg,
    const float* e_Wv, const float* e_bv,
    const float* wslot, const int* cnt, const int* off, float* contrib) {
  int e = blockIdx.z;
  int cnte = cnt[e];
  int m0 = blockIdx.y * 128;
  if (m0 >= cnte) return;
  __shared__ __bf16 sA[128 * SA_LD];
  __shared__ __bf16 sBgT[64 * SBT_LD];
  __shared__ __bf16 sBvT[64 * SBT_LD];
  int tid = threadIdx.x, lane = tid & 31, w = tid >> 5;
  int n0 = blockIdx.x * 64;
  int base = off[e];
  const us* Ae = h1 + (size_t)base * NHID;
  const float* Bg = e_Wg + (size_t)e * NHID * DMODEL;
  const float* Bv = e_Wv + (size_t)e * NHID * DMODEL;
  v8f g0 = {}, g1 = {}, g2 = {}, g3 = {};
  v8f v0 = {}, v1 = {}, v2 = {}, v3 = {};
  for (int kk = 0; kk < NHID; kk += 32) {
    stageA128(sA, Ae, NHID, m0, kk, nullptr, cnte, tid);
    stageBT(sBgT, Bg, DMODEL, n0, kk, tid);
    stageBT(sBvT, Bv, DMODEL, n0, kk, tid);
    if (kk + 32 < NHID) {
      __builtin_prefetch(Bg + (size_t)(kk + 32 + ((tid >> 4) * 2)) * DMODEL + n0, 0, 1);
      __builtin_prefetch(Bv + (size_t)(kk + 32 + ((tid >> 4) * 2)) * DMODEL + n0, 0, 1);
    }
    __syncthreads();
    v16bf af = frag_a(sA + w * 16 * SA_LD, lane);   // load A fragment once, reuse 8x
    mac4(af, sBgT, lane, g0, g1, g2, g3);
    mac4(af, sBvT, lane, v0, v1, v2, v3);
    __syncthreads();
  }
  v8f gaccs[4] = {g0, g1, g2, g3};
  v8f vaccs[4] = {v0, v1, v2, v3};
#pragma unroll
  for (int t = 0; t < 4; ++t) {
#pragma unroll
    for (int i = 0; i < 8; ++i) {
      int rl = w * 16 + i + ((lane >> 4) << 3);
      int cl = t * 16 + (lane & 15);
      int slot = m0 + rl;
      if (slot < cnte) {
        int gcol = n0 + cl;
        float gv = gaccs[t][i] + e_bg[e * DMODEL + gcol];
        float sg = gv / (1.0f + __expf(-gv));
        float vv = vaccs[t][i] + e_bv[e * DMODEL + gcol];
        contrib[(size_t)(base + slot) * DMODEL + gcol] = wslot[e * NT + slot] * sg * vv;
      }
    }
  }
}

// ---------------- final gather + residual ----------------
__global__ __launch_bounds__(256) void k_final(const float* x1, const float* contrib,
                                               const int* eids, const int* posOf, const int* off,
                                               float* out) {
  int t = blockIdx.x;
  int r0 = off[eids[2 * t]]     + posOf[2 * t];
  int r1 = off[eids[2 * t + 1]] + posOf[2 * t + 1];
  for (int c = threadIdx.x; c < DMODEL; c += 256) {
    out[(size_t)t * DMODEL + c] = x1[(size_t)t * DMODEL + c]
        + contrib[(size_t)r0 * DMODEL + c] + contrib[(size_t)r1 * DMODEL + c];
  }
}

// ---------------- host-side orchestration ----------------
extern "C" void kernel_launch(void* const* d_in, const int* in_sizes, int n_in,
                              void* d_out, int out_size, void* d_ws, size_t ws_size,
                              hipStream_t stream) {
  (void)in_sizes; (void)n_in; (void)out_size; (void)ws_size;
  const float* x      = (const float*)d_in[0];
  const float* fcos   = (const float*)d_in[1];
  const float* fsin   = (const float*)d_in[2];
  const int*   tids   = (const int*)  d_in[3];
  const float* norm1w = (const float*)d_in[4];
  const float* norm2w = (const float*)d_in[5];
  const float* WqT    = (const float*)d_in[6];
  const float* WkT    = (const float*)d_in[7];
  const float* WvT    = (const float*)d_in[8];
  const float* WoT    = (const float*)d_in[9];
  const float* qA = (const float*)d_in[10], *qB = (const float*)d_in[11];
  const float* kA = (const float*)d_in[12], *kB = (const float*)d_in[13];
  const float* vA = (const float*)d_in[14], *vB = (const float*)d_in[15];
  const float* gateWT = (const float*)d_in[16];
  const float* gA = (const float*)d_in[17], *gB = (const float*)d_in[18];
  const float* task_emb = (const float*)d_in[19];
  const float* e_W1 = (const float*)d_in[20], *e_b1 = (const float*)d_in[21];
  const float* e_Wg = (const float*)d_in[22], *e_bg = (const float*)d_in[23];
  const float* e_Wv = (const float*)d_in[24], *e_bv = (const float*)d_in[25];
  float* outp = (float*)d_out;

  char* p = (char*)d_ws;
  auto carve = [&](size_t bytes) -> char* {
    char* r = p;
    p += (bytes + 255) & ~(size_t)255;
    return r;
  };
  float* Weff   = (float*)carve((size_t)DMODEL * QKVW * 4);
  float* geff   = (float*)carve((size_t)DMODEL * NE * 4);
  us*    h_ln   = (us*)   carve((size_t)NT * DMODEL * 2);
  float* QKV    = (float*)carve((size_t)NT * QKVW * 4);
  us*    Qr     = (us*)   carve((size_t)NT * DMODEL * 2);
  us*    Kr     = (us*)   carve((size_t)NT * KVD * 2);
  us*    Vr     = (us*)   carve((size_t)NT * KVD * 2);
  us*    ctx    = (us*)   carve((size_t)NT * DMODEL * 2);
  float* x1     = (float*)carve((size_t)NT * DMODEL * 4);
  us*    xf     = (us*)   carve((size_t)NT * DMODEL * 2);
  int*   eids   = (int*)  carve((size_t)NT * 2 * 4);
  float* gwts   = (float*)carve((size_t)NT * 2 * 4);
  int*   posOf  = (int*)  carve((size_t)NT * 2 * 4);
  int*   counts = (int*)  carve((size_t)NE * 4);
  int*   offs   = (int*)  carve((size_t)(NE + 1) * 4);
  int*   perm   = (int*)  carve((size_t)NE * NT * 4);
  float* wslot  = (float*)carve((size_t)NE * NT * 4);
  us*    h1     = (us*)   carve((size_t)NT * 2 * NHID * 2);
  float* contrib= (float*)carve((size_t)NT * 2 * DMODEL * 4);

  // 1) fold LoRA into effective weights
  k_prep_weff<<<(DMODEL * QKVW) / 256, 256, 0, stream>>>(WqT, WkT, WvT, qA, qB, kA, kB, vA, vB, Weff);
  k_prep_geff<<<(DMODEL * NE) / 256, 256, 0, stream>>>(gateWT, gA, gB, geff);
  // 2) rmsnorm1 -> bf16
  k_rmsnorm<<<NT, 256, 0, stream>>>(x, norm1w, nullptr, nullptr, h_ln);
  // 3) fused QKV GEMM (WMMA bf16), 128-row M tiles
  k_gemm_plain<<<dim3(QKVW / 64, NT / 128), 256, 0, stream>>>(h_ln, DMODEL, Weff, QKVW,
                                                              nullptr, QKV, nullptr, QKVW, DMODEL);
  // 4) RoPE + head-major repack -> bf16
  k_rope<<<NT, 256, 0, stream>>>(QKV, fcos, fsin, Qr, Kr, Vr);
  // 5) causal GQA flash attention (WMMA)
  k_flash<<<dim3(SEQ / 128, NH, NB), 256, 0, stream>>>(Qr, Kr, Vr, ctx);
  // 6) output projection + residual: x1 = x + ctx @ WoT
  k_gemm_plain<<<dim3(DMODEL / 64, NT / 128), 256, 0, stream>>>(ctx, DMODEL, WoT, DMODEL,
                                                                x, x1, nullptr, DMODEL, DMODEL);
  // 7) rmsnorm2 + task embedding -> bf16
  k_rmsnorm<<<NT, 256, 0, stream>>>(x1, norm2w, task_emb, tids, xf);
  // 8) gate + deterministic routing
  k_gate<<<NT / 8, 256, 0, stream>>>(xf, geff, eids, gwts);
  k_build_lists<<<NE, 32, 0, stream>>>(eids, gwts, perm, wslot, posOf, counts);
  k_scan<<<1, 32, 0, stream>>>(counts, offs);
  // 9) routed expert GEMMs (WMMA), 128-row M tiles over each expert's token list
  k_gemm_e1<<<dim3(NHID / 64, NT / 128, NE), 256, 0, stream>>>(xf, e_W1, e_b1, perm, counts, offs, h1);
  k_gemm_e2<<<dim3(DMODEL / 64, NT / 128, NE), 256, 0, stream>>>(h1, e_Wg, e_bg, e_Wv, e_bv,
                                                                 wslot, counts, offs, contrib);
  // 10) gather contributions + residual
  k_final<<<NT, 256, 0, stream>>>(x1, contrib, eids, posOf, offs, outp);
}
